// ProbSparseAttention_21363167330764
// MI455X (gfx1250) — matrix operand order repaired and checked by reading, not compiled, not run on hardware
//
#include <hip/hip_runtime.h>
#include <hip/hip_bf16.h>
#include <math.h>

// Problem constants (fixed by the reference)
constexpr int B  = 2, H = 8, L = 2048, S = 2048, E = 64, U = 2048;
constexpr int BH = B * H;

typedef float        v2f   __attribute__((ext_vector_type(2)));
typedef float        v8f   __attribute__((ext_vector_type(8)));
typedef __bf16       v16bf __attribute__((ext_vector_type(16)));
typedef unsigned int v4u   __attribute__((ext_vector_type(4)));

union Frag16 { v16bf v; v4u q[2]; };   // ext-vector members: trivially constructible

// ---------------------------------------------------------------------------
// K1: sparsity measure m[b,h,l] = max_s(qk) - mean_s(qk), qk = q_smp @ k_smpT
// f32 WMMA 16x16x4, 16 k-steps over E=64. Block = 8 waves = 16 rows x 2048 cols.
// ---------------------------------------------------------------------------
__global__ __launch_bounds__(256) void k1_sample_measure(
    const float* __restrict__ q, const float* __restrict__ km,
    const int* __restrict__ ridx, float* __restrict__ mout) {
  __shared__ float qt[16][E + 4];
  __shared__ float kt[8][16][E + 4];
  __shared__ float pmax[16][128];
  __shared__ float psum[16][128];

  const int bh  = blockIdx.y;
  const int r0  = blockIdx.x * 16;
  const int tid = threadIdx.x;
  const int w = tid >> 5, lane = tid & 31, hn = lane & 15, half = lane >> 4;
  const int*   ri = ridx + (size_t)bh * L;
  const float* qb = q  + (size_t)bh * L * E;
  const float* kb = km + (size_t)bh * S * E;

  for (int i = tid; i < 16 * E; i += 256) {
    const int row = i >> 6, e = i & 63;
    qt[row][e] = qb[(size_t)ri[r0 + row] * E + e];
  }
  __syncthreads();

  float lmax[8], lsum[8];
#pragma unroll
  for (int r = 0; r < 8; ++r) { lmax[r] = -INFINITY; lsum[r] = 0.f; }

  for (int it = 0; it < L / (8 * 16); ++it) {
    const int c0 = (it * 8 + w) * 16;
#pragma unroll
    for (int jj = 0; jj < 8; ++jj) {
      const int i = lane + 32 * jj;
      const int row = i >> 4, v4 = i & 15;
      const float4 val = *(const float4*)(kb + (size_t)ri[c0 + row] * E + v4 * 4);
      *(float4*)(&kt[w][row][v4 * 4]) = val;
    }
    __syncthreads();

    v8f acc = {};
#pragma unroll
    for (int kk = 0; kk < E / 4; ++kk) {
      const int e0 = 4 * kk + 2 * half;
      const v2f a = *(const v2f*)&qt[hn][e0];
      const v2f b = *(const v2f*)&kt[w][hn][e0];
      acc = __builtin_amdgcn_wmma_f32_16x16x4_f32(false, a, false, b,
                                                  (short)0, acc, false, false);
    }
#pragma unroll
    for (int r = 0; r < 8; ++r) { lmax[r] = fmaxf(lmax[r], acc[r]); lsum[r] += acc[r]; }
    __syncthreads();
  }

#pragma unroll
  for (int r = 0; r < 8; ++r) {
    pmax[r + 8 * half][w * 16 + hn] = lmax[r];
    psum[r + 8 * half][w * 16 + hn] = lsum[r];
  }
  __syncthreads();
  if (tid < 16) {
    float mx = -INFINITY, sm = 0.f;
    for (int j = 0; j < 128; ++j) { mx = fmaxf(mx, pmax[tid][j]); sm += psum[tid][j]; }
    mout[(size_t)bh * L + r0 + tid] = mx - sm * (1.0f / (float)L);
  }
}

// ---------------------------------------------------------------------------
// K2: per-(b,h) argsort of m descending (U == L, so top_k == full sort).
// ---------------------------------------------------------------------------
__global__ __launch_bounds__(1024) void k2_argsort_desc(
    const float* __restrict__ m, int* __restrict__ order) {
  __shared__ float sv[2048];
  __shared__ int   si[2048];
  const int bh = blockIdx.x, t = threadIdx.x;
  sv[t]        = m[(size_t)bh * L + t];        si[t]        = t;
  sv[t + 1024] = m[(size_t)bh * L + t + 1024]; si[t + 1024] = t + 1024;
  for (int k = 2; k <= 2048; k <<= 1) {
    for (int j = k >> 1; j > 0; j >>= 1) {
      __syncthreads();
      const int i = ((t & ~(j - 1)) << 1) | (t & (j - 1));
      const int p = i | j;
      const float av = sv[i], bv = sv[p];
      const int   ai = si[i], bi = si[p];
      const bool a_first = (av > bv) || (av == bv && ai < bi);
      const bool up = ((i & k) == 0);
      if (up ? !a_first : a_first) {
        sv[i] = bv; sv[p] = av; si[i] = bi; si[p] = ai;
      }
    }
  }
  __syncthreads();
  order[(size_t)bh * U + t]        = si[t];
  order[(size_t)bh * U + t + 1024] = si[t + 1024];
}

// ---------------------------------------------------------------------------
// K3: scores = q[order] @ k^T (f32 WMMA 16x16x4), f32 scores staged in d_out.
// ---------------------------------------------------------------------------
__global__ __launch_bounds__(256) void k3_scores(
    const float* __restrict__ q, const float* __restrict__ km,
    const int* __restrict__ order, float* __restrict__ scores) {
  __shared__ float qt[16][E + 4];
  __shared__ float kt[8][16][E + 4];

  const int bh  = blockIdx.y;
  const int u0  = blockIdx.x * 16;
  const int tid = threadIdx.x;
  const int w = tid >> 5, lane = tid & 31, hn = lane & 15, half = lane >> 4;
  const int*   og = order + (size_t)bh * U;
  const float* qb = q  + (size_t)bh * L * E;
  const float* kb = km + (size_t)bh * S * E;

  for (int i = tid; i < 16 * E; i += 256) {
    const int row = i >> 6, e = i & 63;
    qt[row][e] = qb[(size_t)og[u0 + row] * E + e];
  }
  __syncthreads();

  for (int it = 0; it < S / (8 * 16); ++it) {
    const int c0 = (it * 8 + w) * 16;
#pragma unroll
    for (int jj = 0; jj < 8; ++jj) {
      const int i = lane + 32 * jj;
      const int row = i >> 4, v4 = i & 15;
      const float4 val = *(const float4*)(kb + (size_t)(c0 + row) * E + v4 * 4);
      *(float4*)(&kt[w][row][v4 * 4]) = val;
    }
    __syncthreads();

    v8f acc = {};
#pragma unroll
    for (int kk = 0; kk < E / 4; ++kk) {
      const int e0 = 4 * kk + 2 * half;
      const v2f a = *(const v2f*)&qt[hn][e0];
      const v2f b = *(const v2f*)&kt[w][hn][e0];
      acc = __builtin_amdgcn_wmma_f32_16x16x4_f32(false, a, false, b,
                                                  (short)0, acc, false, false);
    }
    const int col = c0 + hn;
#pragma unroll
    for (int r = 0; r < 8; ++r)
      scores[((size_t)bh * U + u0 + r + 8 * half) * S + col] = acc[r];
    __syncthreads();
  }
}

// ---------------------------------------------------------------------------
// K3b: softmax row statistics (max, sum of exp) per scores row.
// ---------------------------------------------------------------------------
__global__ __launch_bounds__(256) void k3b_softmax_stats(
    const float* __restrict__ scores, float* __restrict__ rmax,
    float* __restrict__ rsum) {
  __shared__ float red[256];
  const size_t row = blockIdx.x;
  const float* p = scores + row * S;
  const int t = threadIdx.x;
  float mx = -INFINITY;
  for (int i = t; i < S; i += 256) mx = fmaxf(mx, p[i]);
  red[t] = mx; __syncthreads();
  for (int o = 128; o > 0; o >>= 1) {
    if (t < o) red[t] = fmaxf(red[t], red[t + o]);
    __syncthreads();
  }
  const float rm = red[0]; __syncthreads();
  float sm = 0.f;
  for (int i = t; i < S; i += 256) sm += __expf(p[i] - rm);
  red[t] = sm; __syncthreads();
  for (int o = 128; o > 0; o >>= 1) {
    if (t < o) red[t] += red[t + o];
    __syncthreads();
  }
  if (t == 0) { rmax[row] = rm; rsum[row] = red[0]; }
}

// ---------------------------------------------------------------------------
// K3c: build WMMA-ready bf16 operands from f32 scores:
//   attnB[u][s] = bf16( softmax(scores)[u][s] )   (row-major: A-frag pairs packed)
//   scT[s][u]   = bf16( scores[u][s] )            (transposed: B-frag pairs packed)
// 32x32 tiles, transpose through 32x33 LDS.
// ---------------------------------------------------------------------------
__global__ __launch_bounds__(256) void k3c_convert(
    const float* __restrict__ scores, const float* __restrict__ rmax,
    const float* __restrict__ rsum, __hip_bfloat16* __restrict__ attnB,
    __hip_bfloat16* __restrict__ scT) {
  __shared__ float lt[32][33];
  __shared__ float srm[32], sinv[32];
  const int bh = blockIdx.z, u0 = blockIdx.y * 32, s0 = blockIdx.x * 32;
  const int tid = threadIdx.x;
  if (tid < 32) {
    const size_t r = (size_t)bh * U + u0 + tid;
    srm[tid]  = rmax[r];
    sinv[tid] = 1.0f / rsum[r];
  }
  __syncthreads();
  const size_t base = (size_t)bh * U * S;
#pragma unroll
  for (int j = 0; j < 4; ++j) {
    const int r = (tid >> 5) + 8 * j, c = tid & 31;
    const float v = scores[base + (size_t)(u0 + r) * S + s0 + c];
    lt[r][c] = v;
    attnB[base + (size_t)(u0 + r) * S + s0 + c] =
        __float2bfloat16(__expf(v - srm[r]) * sinv[r]);
  }
  __syncthreads();
#pragma unroll
  for (int j = 0; j < 4; ++j) {
    const int rr = (tid >> 5) + 8 * j, cc = tid & 31;
    scT[base + (size_t)(s0 + rr) * U + u0 + cc] = __float2bfloat16(lt[cc][rr]);
  }
}

// ---------------------------------------------------------------------------
// K4: ctx = attnB @ scores (bf16 WMMA 16x16x32, f32 accumulate).
// LDS-free, barrier-free. Wave tile = 32x32 (4 accumulators); per 32-K step:
// 4 A + 4 B global_load_b128 and 4 WMMAs (2 packed loads per matrix op).
// ---------------------------------------------------------------------------
__global__ __launch_bounds__(256) void k4_ctx(
    const __hip_bfloat16* __restrict__ attnB,
    const __hip_bfloat16* __restrict__ scT, float* __restrict__ out) {
  const int bh  = blockIdx.z;
  const int u0  = blockIdx.y * 32;
  const int cb  = blockIdx.x * 256;
  const int tid = threadIdx.x;
  const int w = tid >> 5, lane = tid & 31, hn = lane & 15, half = lane >> 4;
  const int c0 = cb + w * 32;
  const size_t base = (size_t)bh * U * S;   // element offset (bf16 elems)

  // packed-dword (bf16-pair) row pointers
  const unsigned int* arow0 = (const unsigned int*)(attnB + base + (size_t)(u0 + hn) * S);
  const unsigned int* arow1 = (const unsigned int*)(attnB + base + (size_t)(u0 + 16 + hn) * S);
  const unsigned int* brow0 = (const unsigned int*)(scT  + base + (size_t)(c0 + hn) * U);
  const unsigned int* brow1 = (const unsigned int*)(scT  + base + (size_t)(c0 + 16 + hn) * U);

  v8f acc00 = {}, acc01 = {}, acc10 = {}, acc11 = {};
  for (int k0 = 0; k0 < S; k0 += 32) {
    const int kd = k0 >> 1;                 // dword index into packed rows
    Frag16 a0, a1, b0, b1;
    // A 16-bit layout: dword pairs {4h..4h+3} then {8+4h..8+4h+3}
    a0.q[0] = *(const v4u*)(arow0 + kd + 4 * half);
    a0.q[1] = *(const v4u*)(arow0 + kd + 8 + 4 * half);
    a1.q[0] = *(const v4u*)(arow1 + kd + 4 * half);
    a1.q[1] = *(const v4u*)(arow1 + kd + 8 + 4 * half);
    // B 16-bit layout: dword pairs {4h..4h+3} then {8+4h..8+4h+3}
    b0.q[0] = *(const v4u*)(brow0 + kd + 4 * half);
    b0.q[1] = *(const v4u*)(brow0 + kd + 8 + 4 * half);
    b1.q[0] = *(const v4u*)(brow1 + kd + 4 * half);
    b1.q[1] = *(const v4u*)(brow1 + kd + 8 + 4 * half);

    __builtin_prefetch(brow0 + kd + 16, 0, 1);   // next K slab -> global_prefetch
    __builtin_prefetch(brow1 + kd + 16, 0, 1);

    acc00 = __builtin_amdgcn_wmma_f32_16x16x32_bf16(false, a0.v, false, b0.v,
                                                    (short)0, acc00, false, false);
    acc01 = __builtin_amdgcn_wmma_f32_16x16x32_bf16(false, a0.v, false, b1.v,
                                                    (short)0, acc01, false, false);
    acc10 = __builtin_amdgcn_wmma_f32_16x16x32_bf16(false, a1.v, false, b0.v,
                                                    (short)0, acc10, false, false);
    acc11 = __builtin_amdgcn_wmma_f32_16x16x32_bf16(false, a1.v, false, b1.v,
                                                    (short)0, acc11, false, false);
  }

  const size_t orow = (size_t)bh * U;
#pragma unroll
  for (int r = 0; r < 8; ++r) {
    const size_t r0 = (orow + u0 + r + 8 * half) * (size_t)S;
    const size_t r1 = (orow + u0 + 16 + r + 8 * half) * (size_t)S;
    out[r0 + c0 + hn]      = acc00[r];
    out[r0 + c0 + 16 + hn] = acc01[r];
    out[r1 + c0 + hn]      = acc10[r];
    out[r1 + c0 + 16 + hn] = acc11[r];
  }
}

// ---------------------------------------------------------------------------
extern "C" void kernel_launch(void* const* d_in, const int* in_sizes, int n_in,
                              void* d_out, int out_size, void* d_ws, size_t ws_size,
                              hipStream_t stream) {
  (void)in_sizes; (void)n_in; (void)out_size; (void)ws_size;
  const float* q    = (const float*)d_in[0];  // (B,L,H,E) == flat (B,H,L,E) via reshape
  const float* k    = (const float*)d_in[1];  // (B,S,H,E) == flat (B,H,S,E)
  // d_in[2] (values) is dead code in the reference
  const int*   ridx = (const int*)d_in[3];    // (B,H,L)
  float* out = (float*)d_out;

  const size_t elems = (size_t)BH * U * S;    // 67,108,864

  // f32 scores staged in d_out (fully overwritten by K4's final ctx)
  float* scores = out;

  // workspace: bf16 operand copies + small arrays (~269 MB total)
  __hip_bfloat16* ws_attnB = (__hip_bfloat16*)d_ws;          // elems * 2B
  __hip_bfloat16* ws_scT   = ws_attnB + elems;               // elems * 2B
  float* ws_m     = (float*)(ws_scT + elems);                // B*H*L
  int*   ws_order = (int*)(ws_m + (size_t)BH * L);           // B*H*U
  float* ws_rmax  = (float*)(ws_order + (size_t)BH * U);     // B*H*U
  float* ws_rsum  = ws_rmax + (size_t)BH * U;                // B*H*U

  k1_sample_measure<<<dim3(L / 16, BH), 256, 0, stream>>>(q, k, ridx, ws_m);
  k2_argsort_desc  <<<dim3(BH), 1024, 0, stream>>>(ws_m, ws_order);
  k3_scores        <<<dim3(U / 16, BH), 256, 0, stream>>>(q, k, ws_order, scores);
  k3b_softmax_stats<<<dim3(BH * U), 256, 0, stream>>>(scores, ws_rmax, ws_rsum);
  k3c_convert      <<<dim3(S / 32, U / 32, BH), 256, 0, stream>>>(scores, ws_rmax, ws_rsum,
                                                                  ws_attnB, ws_scT);
  k4_ctx           <<<dim3(S / 256, U / 32, BH), 256, 0, stream>>>(ws_attnB, ws_scT, out);
}
